// DAHGCN_85220741087879
// MI455X (gfx1250) — matrix operand order repaired and checked by reading, not compiled, not run on hardware
//
#include <hip/hip_runtime.h>
#include <hip/hip_bf16.h>

#define NN    10000
#define FEAT  512
#define HID   256
#define EMB   128
#define NC    10
#define KTOP  10
#define CTILES (NN / 16)          /* 625 16-col tiles  */
#define CGROUPS (NN / 80)         /* 125 80-col groups */
#define APAD  8                   /* LDS row padding (halves) */
#define AROW  (FEAT + APAD)       /* 520 halves = 1040 B, 260 words (=4 mod 64) */

typedef __attribute__((ext_vector_type(16))) __bf16 v16bf;
typedef __attribute__((ext_vector_type(8)))  float  v8f;
typedef __attribute__((ext_vector_type(4)))  int    v4i;
typedef unsigned short u16;
typedef unsigned int   u32;

typedef __attribute__((address_space(1))) v4i* gv4i_p;   // global (AS1) v4i*
typedef __attribute__((address_space(3))) v4i* lv4i_p;   // LDS    (AS3) v4i*

union FragBF { v16bf v; uint4 u4[2]; };

#if defined(__has_builtin)
# if __has_builtin(__builtin_amdgcn_global_load_async_to_lds_b128)
#  define HAVE_ASYNC_LDS 1
# endif
#endif
#ifndef HAVE_ASYNC_LDS
# define HAVE_ASYNC_LDS 0
#endif

static __device__ __forceinline__ u16 f2bf(float f) {
    u32 u = __float_as_uint(f);
    u32 r = (u + 0x7FFFu + ((u >> 16) & 1u)) >> 16;   // round-to-nearest-even
    return (u16)r;
}

// ---------------------------------------------------------------------------
// Row L2-normalize + bf16 casts of x (one block per row, 128 threads)
// ---------------------------------------------------------------------------
__global__ __launch_bounds__(128) void k_rownorm(const float* __restrict__ x,
                                                 u16* __restrict__ xbf,
                                                 u16* __restrict__ xnbf) {
    __shared__ float red[128];
    const int i = blockIdx.x, t = threadIdx.x;
    const float* row = x + (size_t)i * FEAT;
    float v[4]; float s = 0.f;
    #pragma unroll
    for (int j = 0; j < 4; ++j) { v[j] = row[t + j * 128]; s += v[j] * v[j]; }
    red[t] = s;
    __syncthreads();
    for (int off = 64; off > 0; off >>= 1) {
        if (t < off) red[t] += red[t + off];
        __syncthreads();
    }
    const float inv = 1.0f / fmaxf(sqrtf(red[0]), 1e-12f);
    #pragma unroll
    for (int j = 0; j < 4; ++j) {
        const int c = t + j * 128;
        xbf [(size_t)i * FEAT + c] = f2bf(v[j]);
        xnbf[(size_t)i * FEAT + c] = f2bf(v[j] * inv);
    }
}

// Cast + transpose weight: W[K,Nc] f32 -> Wt[Nc,K] bf16
__global__ void k_castT(const float* __restrict__ W, u16* __restrict__ Wt,
                        int K, int Ncol) {
    const int tid = blockIdx.x * blockDim.x + threadIdx.x;
    if (tid >= K * Ncol) return;
    const int k = tid / Ncol, n = tid - k * Ncol;
    Wt[(size_t)n * K + k] = f2bf(W[tid]);
}

__global__ void k_fill(float* p, float v, int n) {
    const int t = blockIdx.x * blockDim.x + threadIdx.x;
    if (t < n) p[t] = v;
}
__global__ void k_rsqrt(float* p, int n) {
    const int t = blockIdx.x * blockDim.x + threadIdx.x;
    if (t < n) p[t] = rsqrtf(p[t]);
}
__global__ void k_deg_edges(const int* __restrict__ dst, float* deg, int ne) {
    const int e = blockIdx.x * blockDim.x + threadIdx.x;
    if (e < ne) atomicAdd(&deg[dst[e]], 1.0f);
}
__global__ void k_deg_topo(const int* __restrict__ topk, float* deg, int n) {
    const int t = blockIdx.x * blockDim.x + threadIdx.x;
    if (t >= n * (KTOP - 1)) return;
    const int i = t / (KTOP - 1), j = t - i * (KTOP - 1);
    atomicAdd(&deg[topk[i * KTOP + 1 + j]], 1.0f);
}

// ---------------------------------------------------------------------------
// Fused cosine-sim GEMM (bf16 WMMA) + running top-10 per row.
// Block = 16 rows (A panel staged in LDS via async copy), 4 waves sweep
// disjoint quarters of 125 column groups; each group = 16x80 tile built from
// 5 independent WMMA accumulation chains sharing one A fragment.
// ---------------------------------------------------------------------------
__global__ __launch_bounds__(128) void k_knn(const u16* __restrict__ xn,
                                             int* __restrict__ topk) {
    __shared__ __align__(16) u16 aLDS[16][AROW];     // 16 rows x 512 (+pad)
    __shared__ float stile[4][16][5 * 17];
    __shared__ float svals[4][16][KTOP];
    __shared__ int   sidx [4][16][KTOP];

    const int wave = threadIdx.x >> 5;
    const int lane = threadIdx.x & 31;
    const int hl   = lane >> 4;
    const int l15  = lane & 15;
    const int rowbase = blockIdx.x * 16;

    // ---- stage the block's 16 A rows into LDS (16 KB) --------------------
    for (int cidx = threadIdx.x; cidx < 16 * (FEAT / 8); cidx += 128) {
        const int r = cidx >> 6;           // FEAT/8 = 64 chunks of 16 B per row
        const int o = (cidx & 63) * 8;     // halfword offset
        const u16* gsrc = xn + (size_t)(rowbase + r) * FEAT + o;
        u16* ldst = &aLDS[r][o];
#if HAVE_ASYNC_LDS
        __builtin_amdgcn_global_load_async_to_lds_b128(
            (gv4i_p)const_cast<u16*>(gsrc), (lv4i_p)ldst, 0, 0);
#else
        *(uint4*)ldst = *(const uint4*)gsrc;
#endif
    }
#if HAVE_ASYNC_LDS
# if defined(__has_builtin) && __has_builtin(__builtin_amdgcn_s_wait_asynccnt)
    __builtin_amdgcn_s_wait_asynccnt(0);
# else
    asm volatile("s_wait_asynccnt 0" ::: "memory");
# endif
#endif
    __syncthreads();

    // ---- per-wave column-group range -------------------------------------
    const int per = (CGROUPS + 3) / 4;               // 32
    const int g0  = wave * per;
    int g1 = g0 + per; if (g1 > CGROUPS) g1 = CGROUPS;

    float val[KTOP]; int idx[KTOP];
    #pragma unroll
    for (int t = 0; t < KTOP; ++t) { val[t] = -3.0e38f; idx[t] = rowbase + l15; }

    for (int g = g0; g < g1; ++g) {
        const int colbase = g * 80;
        const u16* brow0 = xn + (size_t)(colbase + l15) * FEAT;
        if (g + 1 < g1)
            __builtin_prefetch(xn + (size_t)(colbase + 80 + lane) * FEAT, 0, 1);

        v8f c0 = {}, c1 = {}, c2 = {}, c3 = {}, c4 = {};
        #pragma unroll 2
        for (int k = 0; k < FEAT; k += 32) {
            FragBF a;
            a.u4[0] = *(const uint4*)(&aLDS[l15][k + hl * 16]);        // ds_load_b128
            a.u4[1] = *(const uint4*)(&aLDS[l15][k + hl * 16 + 8]);
            FragBF b0, b1, b2, b3, b4;
            const u16* bk = brow0 + k + hl * 16;
            b0.u4[0] = *(const uint4*)(bk);
            b0.u4[1] = *(const uint4*)(bk + 8);
            b1.u4[0] = *(const uint4*)(bk + 16 * FEAT);
            b1.u4[1] = *(const uint4*)(bk + 16 * FEAT + 8);
            b2.u4[0] = *(const uint4*)(bk + 32 * FEAT);
            b2.u4[1] = *(const uint4*)(bk + 32 * FEAT + 8);
            b3.u4[0] = *(const uint4*)(bk + 48 * FEAT);
            b3.u4[1] = *(const uint4*)(bk + 48 * FEAT + 8);
            b4.u4[0] = *(const uint4*)(bk + 64 * FEAT);
            b4.u4[1] = *(const uint4*)(bk + 64 * FEAT + 8);
            c0 = __builtin_amdgcn_wmma_f32_16x16x32_bf16(false, a.v, false, b0.v, (short)0, c0, false, false);
            c1 = __builtin_amdgcn_wmma_f32_16x16x32_bf16(false, a.v, false, b1.v, (short)0, c1, false, false);
            c2 = __builtin_amdgcn_wmma_f32_16x16x32_bf16(false, a.v, false, b2.v, (short)0, c2, false, false);
            c3 = __builtin_amdgcn_wmma_f32_16x16x32_bf16(false, a.v, false, b3.v, (short)0, c3, false, false);
            c4 = __builtin_amdgcn_wmma_f32_16x16x32_bf16(false, a.v, false, b4.v, (short)0, c4, false, false);
        }
        #pragma unroll
        for (int r = 0; r < 8; ++r) {
            const int m = r + hl * 8;
            stile[wave][m][0 * 17 + l15] = c0[r];
            stile[wave][m][1 * 17 + l15] = c1[r];
            stile[wave][m][2 * 17 + l15] = c2[r];
            stile[wave][m][3 * 17 + l15] = c3[r];
            stile[wave][m][4 * 17 + l15] = c4[r];
        }
        asm volatile("s_wait_dscnt 0" ::: "memory");  // wave-local LDS visibility
        __builtin_amdgcn_wave_barrier();

        if (lane < 16) {
            #pragma unroll 4
            for (int n = 0; n < 80; ++n) {
                const float v = stile[wave][lane][(n >> 4) * 17 + (n & 15)];
                if (v > val[KTOP - 1]) {
                    float cv = v; int ci = colbase + n;
                    #pragma unroll
                    for (int t = 0; t < KTOP; ++t) {
                        if (cv > val[t]) {
                            float tv = val[t]; int ti = idx[t];
                            val[t] = cv; idx[t] = ci; cv = tv; ci = ti;
                        }
                    }
                }
            }
        }
        asm volatile("s_wait_dscnt 0" ::: "memory");
        __builtin_amdgcn_wave_barrier();
    }

    if (lane < 16) {
        #pragma unroll
        for (int t = 0; t < KTOP; ++t) {
            svals[wave][lane][t] = val[t];
            sidx [wave][lane][t] = idx[t];
        }
    }
    __syncthreads();

    if (wave == 0 && lane < 16) {                     // merge 4 partial lists
        float fv[KTOP]; int fi[KTOP];
        #pragma unroll
        for (int t = 0; t < KTOP; ++t) { fv[t] = -3.0e38f; fi[t] = rowbase + lane; }
        for (int w = 0; w < 4; ++w) {
            #pragma unroll
            for (int t = 0; t < KTOP; ++t) {
                float cv = svals[w][lane][t]; int ci = sidx[w][lane][t];
                if (cv > fv[KTOP - 1]) {
                    #pragma unroll
                    for (int u = 0; u < KTOP; ++u) {
                        if (cv > fv[u]) {
                            float tv = fv[u]; int ti = fi[u];
                            fv[u] = cv; fi[u] = ci; cv = tv; ci = ti;
                        }
                    }
                }
            }
        }
        #pragma unroll
        for (int t = 0; t < KTOP; ++t)
            topk[(rowbase + lane) * KTOP + t] = fi[t];
    }
}

// ---------------------------------------------------------------------------
// Generic bf16 WMMA GEMM: C[M,Nc] = A[M,K] @ Bt[Nc,K]^T   (wave per 16x16 tile)
// Two independent accumulation chains over even/odd K-chunks.
// ---------------------------------------------------------------------------
__global__ __launch_bounds__(128) void k_gemm(const u16* __restrict__ A,
                                              const u16* __restrict__ Bt,
                                              float* __restrict__ C,
                                              int M, int K, int Ncol) {
    const int wave = threadIdx.x >> 5;
    const int lane = threadIdx.x & 31;
    const int hl   = lane >> 4;
    const int l15  = lane & 15;
    const int ctiles = Ncol >> 4;
    const int tile = blockIdx.x * 4 + wave;
    if (tile >= (M >> 4) * ctiles) return;
    const int mt = tile / ctiles, nt = tile - mt * ctiles;

    const u16* arow = A  + (size_t)(mt * 16 + l15) * K;
    const u16* brow = Bt + (size_t)(nt * 16 + l15) * K;

    v8f c0 = {}, c1 = {};
    #pragma unroll 2
    for (int k = 0; k < K; k += 64) {
        FragBF a0, b0, a1, b1;
        a0.u4[0] = *(const uint4*)(arow + k + hl * 16);
        a0.u4[1] = *(const uint4*)(arow + k + hl * 16 + 8);
        b0.u4[0] = *(const uint4*)(brow + k + hl * 16);
        b0.u4[1] = *(const uint4*)(brow + k + hl * 16 + 8);
        a1.u4[0] = *(const uint4*)(arow + k + 32 + hl * 16);
        a1.u4[1] = *(const uint4*)(arow + k + 40 + hl * 16);
        b1.u4[0] = *(const uint4*)(brow + k + 32 + hl * 16);
        b1.u4[1] = *(const uint4*)(brow + k + 40 + hl * 16);
        c0 = __builtin_amdgcn_wmma_f32_16x16x32_bf16(false, a0.v, false, b0.v, (short)0, c0, false, false);
        c1 = __builtin_amdgcn_wmma_f32_16x16x32_bf16(false, a1.v, false, b1.v, (short)0, c1, false, false);
    }
    const v8f c = c0 + c1;
    #pragma unroll
    for (int r = 0; r < 8; ++r)
        C[(size_t)(mt * 16 + r + hl * 8) * Ncol + nt * 16 + l15] = c[r];
}

// ---------------------------------------------------------------------------
// Scatter aggregation (edge list) : acc[d] += h[s] * dinv[s]*dinv[d]
// ---------------------------------------------------------------------------
__global__ void k_agg_edges(const float* __restrict__ h, float* __restrict__ acc,
                            const float* __restrict__ dinv,
                            const int* __restrict__ src, const int* __restrict__ dst,
                            int ne, int cshift) {
    const int tid = blockIdx.x * blockDim.x + threadIdx.x;
    const int e = tid >> cshift;
    if (e >= ne) return;
    const int C = 1 << cshift;
    const int f = tid & (C - 1);
    const int s = src[e], d = dst[e];
    const float w = dinv[s] * dinv[d];
    atomicAdd(&acc[(size_t)d * C + f], h[(size_t)s * C + f] * w);
}

// Topology aggregation: node i scatters to its 9 kNN neighbors
__global__ void k_agg_topo(const float* __restrict__ h, float* __restrict__ acc,
                           const float* __restrict__ dinv,
                           const int* __restrict__ topk, int n, int cshift) {
    const int tid = blockIdx.x * blockDim.x + threadIdx.x;
    const int i = tid >> cshift;
    if (i >= n) return;
    const int C = 1 << cshift;
    const int f = tid & (C - 1);
    const float hv = h[(size_t)i * C + f];
    const float di = dinv[i];
    #pragma unroll
    for (int t = 1; t < KTOP; ++t) {
        const int d = topk[i * KTOP + t];
        atomicAdd(&acc[(size_t)d * C + f], hv * di * dinv[d]);
    }
}

// Epilogue 1: out_bf = bf16(relu(acc + h*dinv^2 + b))  (self-loop + bias + relu)
__global__ void k_epi_relu(const float* __restrict__ acc, const float* __restrict__ h,
                           const float* __restrict__ dinv, const float* __restrict__ b,
                           u16* __restrict__ out, int n, int cshift) {
    const int tid = blockIdx.x * blockDim.x + threadIdx.x;
    const int i = tid >> cshift;
    if (i >= n) return;
    const int C = 1 << cshift;
    const int f = tid & (C - 1);
    const float di = dinv[i];
    const float v = acc[tid] + h[tid] * di * di + b[f];
    out[tid] = f2bf(fmaxf(v, 0.0f));
}

// Epilogue 2: out = acc + h*dinv^2 + b
__global__ void k_epi_lin(const float* __restrict__ acc, const float* __restrict__ h,
                          const float* __restrict__ dinv, const float* __restrict__ b,
                          float* __restrict__ out, int n, int cshift) {
    const int tid = blockIdx.x * blockDim.x + threadIdx.x;
    const int i = tid >> cshift;
    if (i >= n) return;
    const int C = 1 << cshift;
    const int f = tid & (C - 1);
    const float di = dinv[i];
    out[tid] = acc[tid] + h[tid] * di * di + b[f];
}

// ---------------------------------------------------------------------------
// Fusion + Student-t soft assignment (block per node)
// ---------------------------------------------------------------------------
__global__ __launch_bounds__(128) void k_fuse_q(const float* __restrict__ zs,
                                                const float* __restrict__ zt,
                                                const float* __restrict__ fw,
                                                const float* __restrict__ cent,
                                                float* __restrict__ zout,
                                                float* __restrict__ qout) {
    __shared__ float zsh[EMB];
    __shared__ float qsh[16];
    const int i = blockIdx.x, t = threadIdx.x;
    const float beta = 1.0f / (1.0f + __expf(-fw[0]));
    const float zv = beta * zs[(size_t)i * EMB + t] + (1.0f - beta) * zt[(size_t)i * EMB + t];
    zout[(size_t)i * EMB + t] = zv;
    zsh[t] = zv;
    __syncthreads();
    if (t < NC) {
        float sq = 0.f;
        #pragma unroll 8
        for (int e = 0; e < EMB; ++e) {
            const float d = zsh[e] - cent[t * EMB + e];
            sq += d * d;
        }
        qsh[t] = 1.0f / (1.0f + sq);
    }
    __syncthreads();
    if (t < NC) {
        float s = 0.f;
        #pragma unroll
        for (int c = 0; c < NC; ++c) s += qsh[c];
        qout[(size_t)i * NC + t] = qsh[t] / s;
    }
}

// ---------------------------------------------------------------------------
extern "C" void kernel_launch(void* const* d_in, const int* in_sizes, int n_in,
                              void* d_out, int out_size, void* d_ws, size_t ws_size,
                              hipStream_t stream) {
    const float* x     = (const float*)d_in[0];
    const int*   eidx  = (const int*)d_in[1];
    const float* W_s1  = (const float*)d_in[2];
    const float* b_s1  = (const float*)d_in[3];
    const float* W_s2  = (const float*)d_in[4];
    const float* b_s2  = (const float*)d_in[5];
    const float* W_t1  = (const float*)d_in[6];
    const float* b_t1  = (const float*)d_in[7];
    const float* W_t2  = (const float*)d_in[8];
    const float* b_t2  = (const float*)d_in[9];
    const float* fw    = (const float*)d_in[10];
    const float* cent  = (const float*)d_in[11];
    const int ne = in_sizes[1] / 2;
    const int* src = eidx;
    const int* dst = eidx + ne;

    float* z_out = (float*)d_out;
    float* q_out = z_out + (size_t)NN * EMB;

    // workspace carve-out (deterministic)
    char* wptr = (char*)d_ws;
    auto carve = [&](size_t bytes) -> void* {
        void* p = (void*)wptr;
        wptr += (bytes + 255) & ~(size_t)255;
        return p;
    };
    u16*   x_bf   = (u16*)  carve((size_t)NN * FEAT * 2);
    u16*   xn_bf  = (u16*)  carve((size_t)NN * FEAT * 2);
    u16*   w1s    = (u16*)  carve((size_t)FEAT * HID * 2);
    u16*   w2s    = (u16*)  carve((size_t)HID * EMB * 2);
    u16*   w1t    = (u16*)  carve((size_t)FEAT * HID * 2);
    u16*   w2t    = (u16*)  carve((size_t)HID * EMB * 2);
    float* dinv_s = (float*)carve((size_t)NN * 4);
    float* dinv_t = (float*)carve((size_t)NN * 4);
    int*   topk   = (int*)  carve((size_t)NN * KTOP * 4);
    float* h1     = (float*)carve((size_t)NN * HID * 4);
    float* acc1   = (float*)carve((size_t)NN * HID * 4);
    u16*   r1bf   = (u16*)  carve((size_t)NN * HID * 2);
    float* h2     = (float*)carve((size_t)NN * EMB * 4);
    float* acc2   = (float*)carve((size_t)NN * EMB * 4);
    float* zsbuf  = (float*)carve((size_t)NN * EMB * 4);
    float* ztbuf  = (float*)carve((size_t)NN * EMB * 4);

    // 1. normalize + bf16 casts
    k_rownorm<<<NN, 128, 0, stream>>>(x, x_bf, xn_bf);
    k_castT<<<(FEAT * HID + 255) / 256, 256, 0, stream>>>(W_s1, w1s, FEAT, HID);
    k_castT<<<(HID * EMB + 255) / 256, 256, 0, stream>>>(W_s2, w2s, HID, EMB);
    k_castT<<<(FEAT * HID + 255) / 256, 256, 0, stream>>>(W_t1, w1t, FEAT, HID);
    k_castT<<<(HID * EMB + 255) / 256, 256, 0, stream>>>(W_t2, w2t, HID, EMB);

    // 2. structural degrees: deg = 1 (self loop) + in-degree, dinv = deg^-1/2
    k_fill<<<(NN + 255) / 256, 256, 0, stream>>>(dinv_s, 1.0f, NN);
    k_deg_edges<<<(ne + 255) / 256, 256, 0, stream>>>(dst, dinv_s, ne);
    k_rsqrt<<<(NN + 255) / 256, 256, 0, stream>>>(dinv_s, NN);

    // 3. fused cosine-sim WMMA GEMM + top-10 (async A-panel staging)
    k_knn<<<CTILES, 128, 0, stream>>>(xn_bf, topk);

    // 4. topology degrees
    k_fill<<<(NN + 255) / 256, 256, 0, stream>>>(dinv_t, 1.0f, NN);
    k_deg_topo<<<(NN * (KTOP - 1) + 255) / 256, 256, 0, stream>>>(topk, dinv_t, NN);
    k_rsqrt<<<(NN + 255) / 256, 256, 0, stream>>>(dinv_t, NN);

    const int gemm1_blocks = (NN / 16) * (HID / 16) / 4;   // 2500
    const int gemm2_blocks = (NN / 16) * (EMB / 16) / 4;   // 1250

    // 5. structural branch
    k_gemm<<<gemm1_blocks, 128, 0, stream>>>(x_bf, w1s, h1, NN, FEAT, HID);
    k_fill<<<(NN * HID + 255) / 256, 256, 0, stream>>>(acc1, 0.0f, NN * HID);
    k_agg_edges<<<((size_t)ne * HID + 255) / 256, 256, 0, stream>>>(h1, acc1, dinv_s, src, dst, ne, 8);
    k_epi_relu<<<(NN * HID + 255) / 256, 256, 0, stream>>>(acc1, h1, dinv_s, b_s1, r1bf, NN, 8);
    k_gemm<<<gemm2_blocks, 128, 0, stream>>>(r1bf, w2s, h2, NN, HID, EMB);
    k_fill<<<(NN * EMB + 255) / 256, 256, 0, stream>>>(acc2, 0.0f, NN * EMB);
    k_agg_edges<<<((size_t)ne * EMB + 255) / 256, 256, 0, stream>>>(h2, acc2, dinv_s, src, dst, ne, 7);
    k_epi_lin<<<(NN * EMB + 255) / 256, 256, 0, stream>>>(acc2, h2, dinv_s, b_s2, zsbuf, NN, 7);

    // 6. topology branch (reuses h1/acc1/r1bf/h2/acc2)
    k_gemm<<<gemm1_blocks, 128, 0, stream>>>(x_bf, w1t, h1, NN, FEAT, HID);
    k_fill<<<(NN * HID + 255) / 256, 256, 0, stream>>>(acc1, 0.0f, NN * HID);
    k_agg_topo<<<(NN * HID + 255) / 256, 256, 0, stream>>>(h1, acc1, dinv_t, topk, NN, 8);
    k_epi_relu<<<(NN * HID + 255) / 256, 256, 0, stream>>>(acc1, h1, dinv_t, b_t1, r1bf, NN, 8);
    k_gemm<<<gemm2_blocks, 128, 0, stream>>>(r1bf, w2t, h2, NN, HID, EMB);
    k_fill<<<(NN * EMB + 255) / 256, 256, 0, stream>>>(acc2, 0.0f, NN * EMB);
    k_agg_topo<<<(NN * EMB + 255) / 256, 256, 0, stream>>>(h2, acc2, dinv_t, topk, NN, 7);
    k_epi_lin<<<(NN * EMB + 255) / 256, 256, 0, stream>>>(acc2, h2, dinv_t, b_t2, ztbuf, NN, 7);

    // 7. fusion + soft assignment
    k_fuse_q<<<NN, 128, 0, stream>>>(zsbuf, ztbuf, fw, cent, z_out, q_out);

    (void)n_in; (void)out_size; (void)ws_size;
}
